// DeepStatisticalSolver_5626407158217
// MI455X (gfx1250) — matrix operand order repaired and checked by reading, not compiled, not run on hardware
//
#include <hip/hip_runtime.h>
#include <cstdint>
#include <cstddef>

// ---------------- problem constants ----------------
#define LATDIM 64
#define TSTEPS 5
#define ALPHA_C 0.001f
#define SLOPE_C 0.2f
#define NBATCH 2
#define NNODES 10000
#define NEDGES 100000

static_assert(NEDGES % 16 == 0, "edge tiles exact");
static_assert(NNODES % 16 == 0, "node tiles exact");

typedef __attribute__((ext_vector_type(16))) _Float16 v16h;
typedef __attribute__((ext_vector_type(8)))  _Float16 v8h;
typedef __attribute__((ext_vector_type(8)))  float    v8f;

// Native fp32 global atomic add (GLOBAL_ATOMIC_ADD_F32) — avoid CAS-loop lowering.
__device__ __forceinline__ void atomAddF32(float* p, float v) {
#if defined(__HIP_DEVICE_COMPILE__)
  unsafeAtomicAdd(p, v);
#else
  atomicAdd(p, v);
#endif
}

// ---------------- WMMA helpers ----------------
__device__ __forceinline__ v16h cat8(v8h lo, v8h hi) {
  v16h r;
#pragma unroll
  for (int i = 0; i < 8; ++i) { r[i] = lo[i]; r[8 + i] = hi[i]; }
  return r;
}

// Y(16 x NQ*16) = X(16 x kpad) * W + bias, via v_wmma_f32_16x16x32_f16.
// Xin: LDS, row-major, 16 rows x ldx halfs.  A-fragment layout per ISA:
//   lane<16 : row=lane,    K = kb+{0..7} and kb+16+{0..7}
//   lane>=16: row=lane-16, K = kb+8+{0..7} and kb+24+{0..7}
// WT: global f16, (NQ*16) rows x kpad halfs; row n holds column n of W.
//   B-fragment: lane n=lane&15, ksel=lane>>4 -> WT[n][kb+ksel*8 + {0..7}] and WT[n][kb+16+ksel*8 + {0..7}].
template<int NQ>
__device__ __forceinline__ void layer4(const _Float16* Xin, int ldx, int kpad,
                                       const _Float16* __restrict__ WT,
                                       const float* __restrict__ bias, int scalarBias,
                                       int lane, v8f* c)
{
  const int n   = lane & 15;
  const int sel = lane >> 4;
#pragma unroll
  for (int q = 0; q < NQ; ++q) {
    float bv = bias[scalarBias ? 0 : (q * 16 + n)];
#pragma unroll
    for (int r = 0; r < 8; ++r) c[q][r] = bv;
  }
  const _Float16* xa = Xin + (size_t)(lane & 15) * ldx + sel * 8;
  for (int kb = 0; kb < kpad; kb += 32) {
    v8h alo = *(const v8h*)(xa + kb);
    v8h ahi = *(const v8h*)(xa + kb + 16);
    v16h a = cat8(alo, ahi);
#pragma unroll
    for (int q = 0; q < NQ; ++q) {
      const _Float16* wr = WT + (size_t)(q * 16 + n) * kpad + sel * 8 + kb;
      v8h blo = *(const v8h*)(wr);
      v8h bhi = *(const v8h*)(wr + 16);
      v16h bm = cat8(blo, bhi);
      c[q] = __builtin_amdgcn_wmma_f32_16x16x32_f16(false, a, false, bm,
                                                    (short)0, c[q], false, false);
    }
  }
}

// C/D layout: element (m = r + 8*sel, n = q*16 + (lane&15)) in c[q][r].
__device__ __forceinline__ void store4(const v8f* c, _Float16* Xout, int lane, bool act)
{
  const int n   = lane & 15;
  const int sel = lane >> 4;
#pragma unroll
  for (int q = 0; q < 4; ++q)
#pragma unroll
    for (int r = 0; r < 8; ++r) {
      float v = c[q][r];
      if (act) v = v > 0.f ? v : v * SLOPE_C;
      Xout[(size_t)(r + 8 * sel) * 64 + q * 16 + n] = (_Float16)v;
    }
}

// ---------------- utility kernels ----------------
__global__ void zero_kernel(float* p, size_t n) {
  size_t i = (size_t)blockIdx.x * blockDim.x + threadIdx.x;
  size_t s = (size_t)gridDim.x * blockDim.x;
  for (; i < n; i += s) p[i] = 0.f;
}

// fp32 W (K x N, row-major) -> f16 W^T (Npad x Kpad, row-major, zero-padded)
__global__ void pack_wt(const float* __restrict__ W, _Float16* __restrict__ WT,
                        int K, int N, int Kpad, int Npad) {
  int idx = blockIdx.x * blockDim.x + threadIdx.x;
  int total = Kpad * Npad;
  if (idx >= total) return;
  int nrow = idx / Kpad;
  int k    = idx % Kpad;
  float v = (nrow < N && k < K) ? W[(size_t)k * N + nrow] : 0.f;
  WT[idx] = (_Float16)v;
}

// preds[0] = dec0(0) + initial_U  (constant across nodes/batches: W@0 = 0)
__global__ __launch_bounds__(256) void pred0_kernel(
    const float* b0, const float* W1, const float* b1,
    const float* W2, const float* b2, const float* W3, const float* b3,
    const float* initU, float* out, int n)
{
  __shared__ float h1[64], h2[64], h3[64];
  __shared__ float cval;
  int t = threadIdx.x;
  if (t < 64) { float v = b0[t]; h1[t] = v > 0.f ? v : v * SLOPE_C; }
  __syncthreads();
  if (t < 64) { float s = b1[t]; for (int i = 0; i < 64; ++i) s += h1[i] * W1[i * 64 + t];
                h2[t] = s > 0.f ? s : s * SLOPE_C; }
  __syncthreads();
  if (t < 64) { float s = b2[t]; for (int i = 0; i < 64; ++i) s += h2[i] * W2[i * 64 + t];
                h3[t] = s > 0.f ? s : s * SLOPE_C; }
  __syncthreads();
  if (t == 0) { float s = b3[0]; for (int i = 0; i < 64; ++i) s += h3[i] * W3[i];
                cval = s + initU[0]; }
  __syncthreads();
  for (int i = blockIdx.x * blockDim.x + t; i < n; i += gridDim.x * blockDim.x)
    out[i] = cval;
}

// ---------------- edge kernel ----------------
struct EdgeArgs {
  const float* A;       // (2, E, 4)
  const float* H;       // (2, N, 64)
  const float* A_mean;  // (4)
  const float* A_std;   // (4)
  float* sf; float* st; float* sl;     // each (2, N, 64)
  const _Float16* wt[3][4];            // [from,to,loop][layer]
  const float*    bs[3][4];
};

__global__ __launch_bounds__(32) void edge_kernel(EdgeArgs ea)
{
  const int lane = threadIdx.x;
  const int tpb  = NEDGES / 16;
  const int bb   = blockIdx.x / tpb;
  const int e0   = (blockIdx.x % tpb) * 16;

  __shared__ _Float16 X0[16 * 160];   // phi_in, padded 130 -> 160
  __shared__ _Float16 X1[16 * 64];
  __shared__ _Float16 X2[16 * 64];
  __shared__ int   sFrom[16], sTo[16];
  __shared__ float sMask[16];

  if (lane < 16) {
    const float* ar = ea.A + ((size_t)bb * NEDGES + e0 + lane) * 4;
    float4 av = *(const float4*)ar;
    int f = (int)av.x, t = (int)av.y;
    sFrom[lane] = f; sTo[lane] = t;
    sMask[lane] = (f == t) ? 1.f : 0.f;
    _Float16* xr = X0 + (size_t)lane * 160;
    xr[128] = (_Float16)((av.z - ea.A_mean[2]) / ea.A_std[2]);
    xr[129] = (_Float16)((av.w - ea.A_mean[3]) / ea.A_std[3]);
#pragma unroll
    for (int k = 130; k < 160; ++k) xr[k] = (_Float16)0.f;
  }
  __syncthreads();

  {   // gather Hf (cols 0..63) and Ht (cols 64..127)
    int m = lane & 15, half = lane >> 4;
    const float* hf = ea.H + ((size_t)bb * NNODES + sFrom[m]) * 64 + half * 32;
    const float* ht = ea.H + ((size_t)bb * NNODES + sTo[m])   * 64 + half * 32;
    _Float16* xf = X0 + (size_t)m * 160 +       half * 32;
    _Float16* xt = X0 + (size_t)m * 160 + 64 +  half * 32;
#pragma unroll
    for (int j = 0; j < 8; ++j) {
      float4 v = ((const float4*)hf)[j];
      xf[4*j+0] = (_Float16)v.x; xf[4*j+1] = (_Float16)v.y;
      xf[4*j+2] = (_Float16)v.z; xf[4*j+3] = (_Float16)v.w;
      float4 w = ((const float4*)ht)[j];
      xt[4*j+0] = (_Float16)w.x; xt[4*j+1] = (_Float16)w.y;
      xt[4*j+2] = (_Float16)w.z; xt[4*j+3] = (_Float16)w.w;
    }
  }
  __syncthreads();

  for (int mlp = 0; mlp < 3; ++mlp) {
    { v8f c[4]; layer4<4>(X0, 160, 160, ea.wt[mlp][0], ea.bs[mlp][0], 0, lane, c);
      store4(c, X1, lane, true); }  __syncthreads();
    { v8f c[4]; layer4<4>(X1,  64,  64, ea.wt[mlp][1], ea.bs[mlp][1], 0, lane, c);
      store4(c, X2, lane, true); }  __syncthreads();
    { v8f c[4]; layer4<4>(X2,  64,  64, ea.wt[mlp][2], ea.bs[mlp][2], 0, lane, c);
      store4(c, X1, lane, true); }  __syncthreads();
    { v8f c[4]; layer4<4>(X1,  64,  64, ea.wt[mlp][3], ea.bs[mlp][3], 0, lane, c);
      float* dst = ((mlp == 0) ? ea.sf : (mlp == 1) ? ea.st : ea.sl)
                   + (size_t)bb * NNODES * 64;
      const int n = lane & 15, sel = lane >> 4;
#pragma unroll
      for (int q = 0; q < 4; ++q)
#pragma unroll
        for (int r = 0; r < 8; ++r) {
          int m = r + 8 * sel;
          float fac = (mlp == 2) ? sMask[m] : (1.f - sMask[m]);
          int node  = (mlp == 0) ? sFrom[m] : sTo[m];
          atomAddF32(dst + (size_t)node * 64 + q * 16 + n, c[q][r] * fac);
        }
      __syncthreads();
    }
  }
}

// ---------------- node kernel ----------------
struct NodeArgs {
  const float* Bfeat;   // (2, N, 2)
  const float* B_mean; const float* B_std;
  float* H;             // (2, N, 64) updated in place
  const float* sf; const float* st; const float* sl;
  const _Float16* wpsi[4]; const float* bpsi[4];
  const _Float16* wdec[4]; const float* bdec[4];
  const float* initU;
  float* outPred;       // (2, N) slice for preds[t+1]
};

__global__ __launch_bounds__(32) void node_kernel(NodeArgs na)
{
  const int lane = threadIdx.x;
  const int tpb  = NNODES / 16;
  const int bb   = blockIdx.x / tpb;
  const int n0   = (blockIdx.x % tpb) * 16;

  __shared__ _Float16 X0[16 * 288];   // psi_in, padded 258 -> 288
  __shared__ _Float16 X1[16 * 64];
  __shared__ _Float16 X2[16 * 64];

  {   // psi_in = [H | sf | st | sl | b]
    int m = lane & 15, half = lane >> 4;
    size_t rowbase = ((size_t)bb * NNODES + n0 + m) * 64 + half * 32;
    const float* srcs[4] = { na.H + rowbase, na.sf + rowbase,
                             na.st + rowbase, na.sl + rowbase };
#pragma unroll
    for (int s = 0; s < 4; ++s) {
      _Float16* xr = X0 + (size_t)m * 288 + s * 64 + half * 32;
#pragma unroll
      for (int j = 0; j < 8; ++j) {
        float4 v = ((const float4*)srcs[s])[j];
        xr[4*j+0] = (_Float16)v.x; xr[4*j+1] = (_Float16)v.y;
        xr[4*j+2] = (_Float16)v.z; xr[4*j+3] = (_Float16)v.w;
      }
    }
    if (lane < 16) {
      const float* br = na.Bfeat + ((size_t)bb * NNODES + n0 + lane) * 2;
      _Float16* xr = X0 + (size_t)lane * 288;
      xr[256] = (_Float16)((br[0] - na.B_mean[0]) / na.B_std[0]);
      xr[257] = (_Float16)((br[1] - na.B_mean[1]) / na.B_std[1]);
    } else {
      _Float16* xr = X0 + (size_t)(lane - 16) * 288;
#pragma unroll
      for (int k = 258; k < 288; ++k) xr[k] = (_Float16)0.f;
    }
  }
  __syncthreads();

  // psi MLP
  { v8f c[4]; layer4<4>(X0, 288, 288, na.wpsi[0], na.bpsi[0], 0, lane, c);
    store4(c, X1, lane, true); }  __syncthreads();
  { v8f c[4]; layer4<4>(X1,  64,  64, na.wpsi[1], na.bpsi[1], 0, lane, c);
    store4(c, X2, lane, true); }  __syncthreads();
  { v8f c[4]; layer4<4>(X2,  64,  64, na.wpsi[2], na.bpsi[2], 0, lane, c);
    store4(c, X1, lane, true); }  __syncthreads();
  { v8f c[4]; layer4<4>(X1,  64,  64, na.wpsi[3], na.bpsi[3], 0, lane, c);
    const int n = lane & 15, sel = lane >> 4;
#pragma unroll
    for (int q = 0; q < 4; ++q)
#pragma unroll
      for (int r = 0; r < 8; ++r) {
        int m = r + 8 * sel;
        size_t gi = ((size_t)bb * NNODES + n0 + m) * 64 + q * 16 + n;
        float hnew = na.H[gi] + ALPHA_C * c[q][r];   // fp32 H + alpha*psi
        na.H[gi] = hnew;
        X2[(size_t)m * 64 + q * 16 + n] = (_Float16)hnew;
      }
    __syncthreads();
  }

  // dec MLP on updated H -> preds[t+1]
  { v8f c[4]; layer4<4>(X2, 64, 64, na.wdec[0], na.bdec[0], 0, lane, c);
    store4(c, X1, lane, true); }  __syncthreads();
  { v8f c[4]; layer4<4>(X1, 64, 64, na.wdec[1], na.bdec[1], 0, lane, c);
    store4(c, X2, lane, true); }  __syncthreads();
  { v8f c[4]; layer4<4>(X2, 64, 64, na.wdec[2], na.bdec[2], 0, lane, c);
    store4(c, X1, lane, true); }  __syncthreads();
  { v8f c[1]; layer4<1>(X1, 64, 64, na.wdec[3], na.bdec[3], 1, lane, c);
    const int n = lane & 15, sel = lane >> 4;
    if (n == 0) {
      float iu = na.initU[0];
#pragma unroll
      for (int r = 0; r < 8; ++r) {
        int m = r + 8 * sel;
        na.outPred[(size_t)bb * NNODES + n0 + m] = c[0][r] + iu;
      }
    }
  }
}

// ---------------- host launcher ----------------
// d_in layout assumption (jax pytree flatten, dict values in insertion order,
// params dict by sorted key: dec, phi_from, phi_loop, phi_to, psi; each MLP
// gives leaves W0,b0,W1,b1,W2,b2,W3,b3):
//   0:A  1:B  2..49:dec[6]  50..89:phi_from[5]  90..129:phi_loop[5]
//   130..169:phi_to[5]  170..209:psi[5]
//   210:A_mean 211:A_std 212:B_mean 213:B_std 214:initial_U
extern "C" void kernel_launch(void* const* d_in, const int* in_sizes, int n_in,
                              void* d_out, int out_size, void* d_ws, size_t ws_size,
                              hipStream_t stream)
{
  (void)in_sizes; (void)n_in; (void)out_size; (void)ws_size;

  const float* A      = (const float*)d_in[0];
  const float* Bfeat  = (const float*)d_in[1];
  const int IDEC = 2, IPF = 50, IPL = 90, IPT = 130, IPSI = 170;
  const float* A_mean = (const float*)d_in[210];
  const float* A_std  = (const float*)d_in[211];
  const float* B_mean = (const float*)d_in[212];
  const float* B_std  = (const float*)d_in[213];
  const float* initU  = (const float*)d_in[214];
  float* out = (float*)d_out;

  // ---- workspace carve-out (~22 MB total) ----
  char* wbase = (char*)d_ws;
  size_t off = 0;
  auto walloc = [&](size_t bytes) -> void* {
    void* p = wbase + off;
    off = (off + bytes + 255) & ~(size_t)255;
    return p;
  };
  const size_t HN = (size_t)NBATCH * NNODES * 64;
  float* H  = (float*)walloc(HN * 4);
  float* sf = (float*)walloc(3 * HN * 4);   // sf|st|sl contiguous
  float* st = sf + HN;
  float* sl = st + HN;

  // ---- pack all weights: fp32 W(KxN) -> f16 W^T(Npad x Kpad) ----
  auto packMat = [&](int leaf, int K, int N, int Kpad, int Npad) -> const _Float16* {
    _Float16* dst = (_Float16*)walloc((size_t)Kpad * Npad * 2);
    int total = Kpad * Npad;
    pack_wt<<<(total + 255) / 256, 256, 0, stream>>>(
        (const float*)d_in[leaf], dst, K, N, Kpad, Npad);
    return dst;
  };

  const _Float16* wdec[6][4]; const float* bdec[6][4];
  for (int i = 0; i < 6; ++i)
    for (int l = 0; l < 4; ++l) {
      int li = IDEC + 8 * i + 2 * l;
      int N = (l == 3) ? 1 : 64, Npad = (l == 3) ? 16 : 64;
      wdec[i][l] = packMat(li, 64, N, 64, Npad);
      bdec[i][l] = (const float*)d_in[li + 1];
    }

  const int phiBase[3] = { IPF, IPT, IPL };   // order: from, to, loop
  const _Float16* wphi[3][TSTEPS][4]; const float* bphi[3][TSTEPS][4];
  for (int g = 0; g < 3; ++g)
    for (int t = 0; t < TSTEPS; ++t)
      for (int l = 0; l < 4; ++l) {
        int li = phiBase[g] + 8 * t + 2 * l;
        int K = (l == 0) ? 130 : 64, Kpad = (l == 0) ? 160 : 64;
        wphi[g][t][l] = packMat(li, K, 64, Kpad, 64);
        bphi[g][t][l] = (const float*)d_in[li + 1];
      }

  const _Float16* wpsi[TSTEPS][4]; const float* bpsi[TSTEPS][4];
  for (int t = 0; t < TSTEPS; ++t)
    for (int l = 0; l < 4; ++l) {
      int li = IPSI + 8 * t + 2 * l;
      int K = (l == 0) ? 258 : 64, Kpad = (l == 0) ? 288 : 64;
      wpsi[t][l] = packMat(li, K, 64, Kpad, 64);
      bpsi[t][l] = (const float*)d_in[li + 1];
    }

  // ---- H = 0; preds[0] = dec0(0) + initU ----
  zero_kernel<<<2048, 256, 0, stream>>>(H, HN);
  pred0_kernel<<<80, 256, 0, stream>>>(
      (const float*)d_in[IDEC + 1], (const float*)d_in[IDEC + 2],
      (const float*)d_in[IDEC + 3], (const float*)d_in[IDEC + 4],
      (const float*)d_in[IDEC + 5], (const float*)d_in[IDEC + 6],
      (const float*)d_in[IDEC + 7], initU, out, NBATCH * NNODES);

  // ---- T iterations ----
  for (int t = 0; t < TSTEPS; ++t) {
    zero_kernel<<<2048, 256, 0, stream>>>(sf, 3 * HN);

    EdgeArgs ea;
    ea.A = A; ea.H = H; ea.A_mean = A_mean; ea.A_std = A_std;
    ea.sf = sf; ea.st = st; ea.sl = sl;
    for (int g = 0; g < 3; ++g)
      for (int l = 0; l < 4; ++l) { ea.wt[g][l] = wphi[g][t][l]; ea.bs[g][l] = bphi[g][t][l]; }
    edge_kernel<<<NBATCH * (NEDGES / 16), 32, 0, stream>>>(ea);

    NodeArgs na;
    na.Bfeat = Bfeat; na.B_mean = B_mean; na.B_std = B_std;
    na.H = H; na.sf = sf; na.st = st; na.sl = sl;
    for (int l = 0; l < 4; ++l) {
      na.wpsi[l] = wpsi[t][l];     na.bpsi[l] = bpsi[t][l];
      na.wdec[l] = wdec[t + 1][l]; na.bdec[l] = bdec[t + 1][l];
    }
    na.initU = initU;
    na.outPred = out + (size_t)(t + 1) * NBATCH * NNODES;
    node_kernel<<<NBATCH * (NNODES / 16), 32, 0, stream>>>(na);
  }
}
